// RowLSTM_23175643529577
// MI455X (gfx1250) — compile-verified, matrix operand order checked by reading, not run on hardware
//
#include <hip/hip_runtime.h>
#include <stdint.h>
#include <math.h>

typedef __attribute__((ext_vector_type(16))) _Float16 v16h;
typedef __attribute__((ext_vector_type(8)))  float    v8f;
typedef __attribute__((ext_vector_type(4))) unsigned int v4u;
typedef __attribute__((ext_vector_type(8)))  int      v8i;
typedef __attribute__((ext_vector_type(4)))  int      v4i;

#define NB    16    // batch
#define CIN   64
#define HH    64    // image height (scan length)
#define WW    64    // image width
#define HID   128
#define GATES 512   // 4*HID
#define KTOT  512   // 3*128 (state taps) + 2*64 (input taps, tap2 masked off)

// LDS layout -----------------------------------------------------------
#define HT_STRIDE 136                        // halves per padded row (bank spread)
#define HT_ROWS   66                         // rows 0 and 65 are zero halo
#define XT_ROWS   65                         // row 0 is zero halo
#define ZSTR      66                         // f32 z/c row stride (bank spread)
#define HT_BYTES  (HT_ROWS * HT_STRIDE * 2)  // 17952
#define XT_BYTES  (XT_ROWS * HT_STRIDE * 2)  // 17680
#define Z_BYTES   (GATES * ZSTR * 4)         // 135168
#define C_BYTES   (HID * ZSTR * 4)           // 33792
#define BIAS_BYTES (GATES * 4)               // 2048

#define HT_OFF   0
#define XB0_OFF  (HT_BYTES)                  // x row buffer 0 (double buffered)
#define XB1_OFF  (HT_BYTES + XT_BYTES)
#define Z_OFF    (HT_BYTES + 2 * XT_BYTES)
#define C_OFF    (Z_OFF + Z_BYTES)
#define BIAS_OFF (C_OFF + C_BYTES)
#define SMEM_BYTES (BIAS_OFF + BIAS_BYTES)   // 224320

__device__ __forceinline__ v16h ldfrag(const _Float16* p) {
    union { v4u u[2]; v16h h; } f;
    f.u[0] = *(const v4u*)(p);
    f.u[1] = *(const v4u*)(p + 8);
    return f.h;
}

// Manual A-fragment load: pinned in place by volatility, tracked by hand.
// One base address pair + immediate offsets -> no per-fragment address regs.
#define ALOAD(lo, hi, base, byteoff)                                          \
    asm volatile("global_load_b128 %0, %2, off offset:%3\n\t"                 \
                 "global_load_b128 %1, %2, off offset:%4"                     \
                 : "=&v"(lo), "=&v"(hi)                                       \
                 : "v"(base), "n"(byteoff), "n"((byteoff) + 16))

// Consume gate: ties the destination regs so all uses follow the wait.
#define AWAIT(lo, hi, n)                                                      \
    asm volatile("s_wait_loadcnt %2" : "+v"(lo), "+v"(hi) : "n"(n))

__device__ __forceinline__ float sigm(float x) {
    return __builtin_amdgcn_rcpf(1.f + __expf(-x));
}
__device__ __forceinline__ float tanh_fast(float x) {
    return 2.f * sigm(2.f * x) - 1.f;   // safe at +-inf limits
}

// TDM: async-copy one prepacked x row (8192B = 2048 dwords) into LDS,
// inserting 36 dwords of pad every 32 dwords -> 272B (136-half) row stride.
__device__ __forceinline__ void tdm_load_row(const _Float16* src, unsigned lds_byte_off) {
    unsigned long long ga = (unsigned long long)(uintptr_t)src;
    v4u g0;
    g0.x = 1u;                                   // count=1, user descriptor
    g0.y = lds_byte_off;                         // LDS dest byte address
    g0.z = (unsigned)ga;                         // global_addr[31:0]
    g0.w = (unsigned)(ga >> 32) | (2u << 30);    // global_addr[56:32] | type=2
    v8i g1;
    //          data_size=4B   pad_en      interval=32dw  amount=36dw
    g1[0] = (int)((2u << 16) | (1u << 20) | (4u << 22) | (35u << 25));
    g1[1] = (int)(2048u << 16);                  // tensor_dim0 = 2048 (low 16)
    g1[2] = (int)(1u << 16);                     // tensor_dim0 hi=0, tensor_dim1 = 1
    g1[3] = (int)(2048u << 16);                  // tensor_dim1 hi=0, tile_dim0 = 2048
    g1[4] = 0;                                   // tile_dim1/2 unused
    g1[5] = 2048;                                // tensor_dim0_stride = 2048
    g1[6] = 0;
    g1[7] = 0;
    v4i gz4 = {0, 0, 0, 0};
    v8i gz8 = {0, 0, 0, 0, 0, 0, 0, 0};
    __builtin_amdgcn_tensor_load_to_lds(g0, g1, gz4, gz4, gz8, 0);
}

// Pack A = [W_ss taps | W_is taps(0,1)] into row-major f16 [512][512].
__global__ void rowlstm_pack_a(const float* __restrict__ w_is,
                               const float* __restrict__ w_ss,
                               _Float16* __restrict__ Apack) {
    int idx = blockIdx.x * 256 + threadIdx.x;
    if (idx >= GATES * KTOT) return;
    int m = idx >> 9;
    int k = idx & (KTOT - 1);
    float v;
    if (k < 384) {
        int t = k >> 7, c = k & 127;
        v = w_ss[(m * HID + c) * 3 + t];
    } else {
        int kk = k - 384;
        int t = kk >> 6, c = kk & 63;
        v = w_is[(m * CIN + c) * 3 + t];
    }
    Apack[(size_t)m * KTOT + k] = (_Float16)v;
}

// Transpose-convert one x row: xTg[(b*H+row)][w][c] = (f16) x[b][c][row][w]
__global__ void rowlstm_pack_x(const float* __restrict__ x,
                               _Float16* __restrict__ xTg) {
    int br = blockIdx.x;
    int b = br >> 6, row = br & 63;
    _Float16* dst = xTg + (size_t)br * (WW * CIN);
    for (int i = threadIdx.x; i < CIN * WW; i += 256) {
        int c = i >> 6, w = i & 63;
        float v = x[(((size_t)b * CIN + c) * HH + row) * WW + w];
        dst[w * CIN + c] = (_Float16)v;
    }
}

// One block per batch element. 1024 threads = 32 waves.
extern "C" __global__ void __launch_bounds__(1024)
rowlstm_main(const _Float16* __restrict__ Apack,
             const _Float16* __restrict__ xTg,
             const float* __restrict__ b_is,
             const float* __restrict__ b_ss,
             float* __restrict__ out) {
    extern __shared__ char smem[];
    _Float16* hT  = (_Float16*)(smem + HT_OFF);    // [66][136] f16, rows 0/65 = 0
    _Float16* xb0 = (_Float16*)(smem + XB0_OFF);   // [65][136] f16, row 0 = 0
    _Float16* xb1 = (_Float16*)(smem + XB1_OFF);
    float*    zb  = (float*)(smem + Z_OFF);        // [512][66] f32
    float*    cb  = (float*)(smem + C_OFF);        // [128][66] f32
    float*    bias= (float*)(smem + BIAS_OFF);     // [512]

    const int tid  = threadIdx.x;
    const int wave = tid >> 5;
    const int lane = tid & 31;
    const int ncol = lane & 15;            // N within tile
    const int hi   = (lane >> 4) & 1;      // K-half select
    const int b    = blockIdx.x;
    const int m0   = wave * 16;
    const int mrow = m0 + (lane & 15);     // A-matrix row this lane feeds

    // ---- init: zero hT + both x buffers (h0 = 0 + halos), zero c, stage bias ----
    for (int i = tid; i < Z_OFF / 4; i += 1024)
        ((unsigned int*)smem)[i] = 0u;
    for (int i = tid; i < HID * ZSTR; i += 1024) cb[i] = 0.f;
    for (int i = tid; i < GATES; i += 1024) bias[i] = b_is[i] + b_ss[i];
    __syncthreads();

    const _Float16* arow = Apack + (size_t)mrow * KTOT + hi * 16;
    const _Float16* xrow0 = xTg + (size_t)b * HH * (WW * CIN);

    // prologue: async-prefetch row 0 into buffer 0 (single wave drives the TDM)
    if (wave == 0) {
        tdm_load_row(xrow0, XB0_OFF + 2 * HT_STRIDE);   // dest skips zero halo row
    }

    for (int row = 0; row < HH; ++row) {
        const _Float16* xTcur = (row & 1) ? xb1 : xb0;

        if (wave == 0) __builtin_amdgcn_s_wait_tensorcnt(0);
        __syncthreads();   // x row ready for all; prev stage-3 hT writes visible

        // prefetch next row into the other buffer while we do the GEMM
        if (wave == 0 && row + 1 < HH) {
            tdm_load_row(xrow0 + (size_t)(row + 1) * (WW * CIN),
                         (((row + 1) & 1) ? XB1_OFF : XB0_OFF) + 2 * HT_STRIDE);
        }

        // ---- fused GEMM  z[512x64] = A[512x512] * B[512x64] ----
        // A fragments hand-pipelined 2 k-groups deep with manual loadcnt
        // tracking: at fragment s, fragments s+1/s+2 (4 loads) stay in
        // flight -> s_wait_loadcnt 4 (then 2, then 0 at the tail).
        // In-loop LOADcnt is used only by these asm loads.
        v8f acc0 = {}, acc1 = {}, acc2 = {}, acc3 = {};
        v4u alo[3], ahi[3];
        ALOAD(alo[0], ahi[0], arow, 0);
        ALOAD(alo[1], ahi[1], arow, 64);
#pragma unroll
        for (int s = 0; s < 16; ++s) {
            if (s + 2 < 16) ALOAD(alo[(s + 2) % 3], ahi[(s + 2) % 3], arow, (s + 2) * 64);
            if (s <= 13)      AWAIT(alo[s % 3], ahi[s % 3], 4);
            else if (s == 14) AWAIT(alo[s % 3], ahi[s % 3], 2);
            else              AWAIT(alo[s % 3], ahi[s % 3], 0);
            union { v4u u[2]; v16h h; } fu;
            fu.u[0] = alo[s % 3];
            fu.u[1] = ahi[s % 3];
            v16h afA = fu.h;

            const _Float16* base;
            int tap, koff;
            if (s < 12) { tap = s >> 2; koff = (s & 3) * 32; base = hT; }     // state taps 0..2
            else { int ss = s - 12; tap = ss >> 1; koff = (ss & 1) * 32; base = xTcur; } // input taps 0..1
            const _Float16* brow = base + (ncol + tap) * HT_STRIDE + koff + hi * 16;
            v16h b0 = ldfrag(brow);
            v16h b1 = ldfrag(brow + 16 * HT_STRIDE);
            v16h b2 = ldfrag(brow + 32 * HT_STRIDE);
            v16h b3 = ldfrag(brow + 48 * HT_STRIDE);
            acc0 = __builtin_amdgcn_wmma_f32_16x16x32_f16(false, afA, false, b0, (short)0, acc0, false, false);
            acc1 = __builtin_amdgcn_wmma_f32_16x16x32_f16(false, afA, false, b1, (short)0, acc1, false, false);
            acc2 = __builtin_amdgcn_wmma_f32_16x16x32_f16(false, afA, false, b2, (short)0, acc2, false, false);
            acc3 = __builtin_amdgcn_wmma_f32_16x16x32_f16(false, afA, false, b3, (short)0, acc3, false, false);
        }
        // scatter D tiles: lane L, vreg r -> (m0 + r + hi*8, n0 + ncol)
        {
            int mbase = (m0 + hi * 8) * ZSTR + ncol;
#pragma unroll
            for (int r = 0; r < 8; ++r) {
                zb[mbase + r * ZSTR +  0] = acc0[r];
                zb[mbase + r * ZSTR + 16] = acc1[r];
                zb[mbase + r * ZSTR + 32] = acc2[r];
                zb[mbase + r * ZSTR + 48] = acc3[r];
            }
        }
        __syncthreads();

        // ---- LSTM elementwise (8192 cells / 1024 threads) ----
#pragma unroll
        for (int k = 0; k < 8; ++k) {
            int idx = tid + k * 1024;
            int hc = idx >> 6;
            int n  = idx & 63;
            float zi = zb[hc * ZSTR + n]           + bias[hc];
            float zf = zb[(hc + 128) * ZSTR + n]   + bias[hc + 128];
            float zo = zb[(hc + 256) * ZSTR + n]   + bias[hc + 256];
            float zg = zb[(hc + 384) * ZSTR + n]   + bias[hc + 384];
            float ig = sigm(zi);
            float fg = sigm(zf);
            float og = sigm(zo);
            float g  = tanh_fast(zg);
            float c_ = fg * cb[hc * ZSTR + n] + ig * g;
            float h_ = og * tanh_fast(c_);
            cb[hc * ZSTR + n] = c_;
            hT[(n + 1) * HT_STRIDE + hc] = (_Float16)h_;       // next step's B matrix
            out[(((size_t)b * HID + hc) * HH + row) * WW + n] = h_;
        }
        // next-iteration top barrier orders stage-3 writes vs. next GEMM reads
    }
}

extern "C" void kernel_launch(void* const* d_in, const int* in_sizes, int n_in,
                              void* d_out, int out_size, void* d_ws, size_t ws_size,
                              hipStream_t stream) {
    const float* x    = (const float*)d_in[0];
    const float* w_is = (const float*)d_in[1];
    const float* b_is = (const float*)d_in[2];
    const float* w_ss = (const float*)d_in[3];
    const float* b_ss = (const float*)d_in[4];
    float* out = (float*)d_out;

    _Float16* Apack = (_Float16*)d_ws;                                     // 512 KB
    _Float16* xTg   = (_Float16*)((char*)d_ws + (size_t)GATES * KTOT * 2); // 8 MB

    rowlstm_pack_a<<<(GATES * KTOT + 255) / 256, 256, 0, stream>>>(w_is, w_ss, Apack);
    rowlstm_pack_x<<<NB * HH, 256, 0, stream>>>(x, xTg);

    (void)hipFuncSetAttribute((const void*)rowlstm_main,
                              hipFuncAttributeMaxDynamicSharedMemorySize, SMEM_BYTES);
    rowlstm_main<<<NB, 1024, SMEM_BYTES, stream>>>(Apack, xTg, b_is, b_ss, out);
}
